// CTRNN_19748259627232
// MI455X (gfx1250) — matrix operand order repaired and checked by reading, not compiled, not run on hardware
//
#include <hip/hip_runtime.h>

// CTRNN on MI455X (gfx1250, wave32, WMMA).
//   T=512, B=256, I=256, H=1024, ALPHA=0.1
// Phase 1: inp_proj = x @ W_in^T + b_in  -> written into d_out (overwritten later).
// Phase 2: ONE persistent kernel runs all 512 recurrent steps:
//   - W_hh column-slice (32 cols x 1024 K, bf16 = 64KB) resident in LDS per WG
//   - f32 hidden state resident in VGPRs (thread owns fixed (m,n) coords)
//   - bf16 ping-pong h buffers in global for the next step's WMMA A operand
//   - device-wide atomic barrier between steps (32 WGs, L2 atomics)

#define T_SEQ   512
#define N_BATCH 256
#define N_IN    256
#define N_HID   1024
#define ALPHA_F 0.1f

#define STEP_WGS    32                  // persistent grid: one WG per 32 hidden cols
#define COLS_PER_WG (N_HID / STEP_WGS)  // 32
#define LDS_STRIDE  1032                // halves per column: 1024 + 8 pad (516 dwords)

typedef __attribute__((ext_vector_type(16))) __bf16 v16bf;
typedef __attribute__((ext_vector_type(8)))  __bf16 v8bf;
typedef __attribute__((ext_vector_type(8)))  float  v8f;

union FragBF { v16bf v; v8bf h8[2]; };

// ---- fragment layouts (CDNA5 ISA 7.12.2, TN GEMM, row-major over K)
// A (16x32 MxK): lane {mi=lane&15 -> M row, hi=lane>>4}; halves 0..7 = K k0+hi*8..+7,
//                halves 8..15 = K k0+16+hi*8..+7  (two 16B chunks)
// B (32x16 KxN): lane {mi -> N col, hi}; halves 0..15 = same K order as A (two 16B chunks)

__device__ __forceinline__ v8f wmma_bf16(v16bf a, v16bf b, v8f c) {
    return __builtin_amdgcn_wmma_f32_16x16x32_bf16(false, a, false, b, (short)0, c,
                                                   false, false);
}

// ---- phase 1: input projection ---------------------------------------------
// out[m,n] = sum_k x[m,k] * W_in[n,k] + b_in[n];  m in [0,T*B), n in [0,H)
// WG = 256 threads = 8 waves stacked along M; wave tile 16(M) x 64(N).
// grid = (T*B/128, H/64) = (1024, 16)
__global__ __launch_bounds__(256) void ctrnn_inproj(
    const float*  __restrict__ x,
    const __bf16* __restrict__ Wi,
    const float*  __restrict__ bi,
    float*        __restrict__ out) {
    const int lane = threadIdx.x & 31;
    const int wave = threadIdx.x >> 5;
    const int mi = lane & 15, hi = lane >> 4;
    const int m0 = blockIdx.x * 128 + wave * 16;
    const int n0 = blockIdx.y * 64;

    v8f acc[4] = {};
    const float* arow = x + (size_t)(m0 + mi) * N_IN + hi * 8;

#pragma unroll 2
    for (int k = 0; k < N_IN; k += 32) {
        v8f f0 = *(const v8f*)(arow + k);
        v8f f1 = *(const v8f*)(arow + k + 16);
        FragBF a;
#pragma unroll
        for (int i = 0; i < 8; ++i) {
            a.h8[0][i] = (__bf16)f0[i];
            a.h8[1][i] = (__bf16)f1[i];
        }
#pragma unroll
        for (int j = 0; j < 4; ++j) {
            const __bf16* bp = Wi + (size_t)(n0 + j * 16 + mi) * N_IN + k + hi * 8;
            FragBF b;
            b.h8[0] = *(const v8bf*)bp;
            b.h8[1] = *(const v8bf*)(bp + 16);
            acc[j] = wmma_bf16(a.v, b.v, acc[j]);
        }
    }
#pragma unroll
    for (int j = 0; j < 4; ++j) {
        const int n = n0 + j * 16 + mi;
        const float bias = bi[n];
#pragma unroll
        for (int r = 0; r < 8; ++r) {
            const int m = m0 + r + hi * 8;
            out[(size_t)m * N_HID + n] = acc[j][r] + bias;
        }
    }
}

// ---- device-wide barrier for the persistent scan ---------------------------
__device__ __forceinline__ void grid_barrier(unsigned* cnt, unsigned* gen) {
    __syncthreads();
    if (threadIdx.x == 0) {
        __threadfence();  // release this WG's global writes
        unsigned g = __hip_atomic_load(gen, __ATOMIC_RELAXED, __HIP_MEMORY_SCOPE_AGENT);
        unsigned arrived =
            __hip_atomic_fetch_add(cnt, 1u, __ATOMIC_ACQ_REL, __HIP_MEMORY_SCOPE_AGENT);
        if (arrived == STEP_WGS - 1) {
            __hip_atomic_store(cnt, 0u, __ATOMIC_RELAXED, __HIP_MEMORY_SCOPE_AGENT);
            __hip_atomic_fetch_add(gen, 1u, __ATOMIC_RELEASE, __HIP_MEMORY_SCOPE_AGENT);
        } else {
            while (__hip_atomic_load(gen, __ATOMIC_ACQUIRE, __HIP_MEMORY_SCOPE_AGENT) == g) {
                __builtin_amdgcn_s_sleep(1);
            }
        }
        __threadfence();  // acquire other WGs' writes
    }
    __syncthreads();
}

// ---- phase 2: persistent recurrent scan ------------------------------------
// grid = 32 WGs x 256 threads. WG b owns hidden cols [32b, 32b+32).
// Wave tile 32(M) x 32(N): 8 waves cover all 256 batch rows.
// h_f32 state lives in registers; W tile lives in LDS for all 512 steps.
__global__ __launch_bounds__(256) void ctrnn_scan(
    const float*  __restrict__ Whh,   // [H,H] f32
    const float*  __restrict__ bh,    // [H]
    float*        __restrict__ out,   // [T,B,H]: holds u_t, overwritten with h_t
    __bf16*       __restrict__ h_bf0, // ping (zero-initialized here)
    __bf16*       __restrict__ h_bf1, // pong
    unsigned*     __restrict__ bar) { // [2]: counter, generation (pre-zeroed)
    __shared__ __bf16 wlds[COLS_PER_WG * LDS_STRIDE];  // 66 KB

    const int tid  = threadIdx.x;
    const int lane = tid & 31;
    const int wave = tid >> 5;
    const int mi = lane & 15, hi = lane >> 4;
    const int n_base = blockIdx.x * COLS_PER_WG;

    // -- load + convert this WG's W_hh column slice into LDS (once) --
    for (int c = wave; c < COLS_PER_WG; c += 8) {
        const float* src = Whh + (size_t)(n_base + c) * N_HID;
        __bf16* dst = wlds + c * LDS_STRIDE;
        for (int k = lane * 8; k < N_HID; k += 32 * 8) {
            v8f f = *(const v8f*)(src + k);
            v8bf o;
#pragma unroll
            for (int i = 0; i < 8; ++i) o[i] = (__bf16)f[i];
            *(v8bf*)(dst + k) = o;
        }
    }

    // -- zero-init the bf16 ping buffer (disjoint slices across all threads) --
    {
        const int stride = STEP_WGS * 256 * 8;
        for (int i = (blockIdx.x * 256 + tid) * 8; i < N_BATCH * N_HID; i += stride) {
            v8bf z = {};
            *(v8bf*)(h_bf0 + i) = z;
        }
    }
    grid_barrier(&bar[0], &bar[1]);  // init + (local) LDS fill visible everywhere

    const int m0 = wave * 32;  // this wave's batch rows: m0..m0+31
    const float bias0 = bh[n_base + mi];
    const float bias1 = bh[n_base + 16 + mi];

    // f32 hidden state in registers: hreg[mtile][ntile][r]
    float hreg[2][2][8];
#pragma unroll
    for (int jm = 0; jm < 2; ++jm)
#pragma unroll
        for (int jn = 0; jn < 2; ++jn)
#pragma unroll
            for (int r = 0; r < 8; ++r) hreg[jm][jn][r] = 0.0f;

    for (int t = 0; t < T_SEQ; ++t) {
        const __bf16* hin  = (t & 1) ? h_bf1 : h_bf0;
        __bf16*       hout = (t & 1) ? h_bf0 : h_bf1;
        float* out_t = out + (size_t)t * N_BATCH * N_HID;

        v8f acc00 = {}, acc01 = {}, acc10 = {}, acc11 = {};
        const __bf16* arow0 = hin + (size_t)(m0 + mi) * N_HID + hi * 8;
        const __bf16* arow1 = hin + (size_t)(m0 + 16 + mi) * N_HID + hi * 8;

#pragma unroll 2
        for (int k = 0; k < N_HID; k += 32) {
            __builtin_prefetch(arow0 + k + 128, 0, 3);
            __builtin_prefetch(arow1 + k + 128, 0, 3);
            FragBF a0, a1, b0, b1;
            a0.h8[0] = *(const v8bf*)(arow0 + k);
            a0.h8[1] = *(const v8bf*)(arow0 + k + 16);
            a1.h8[0] = *(const v8bf*)(arow1 + k);
            a1.h8[1] = *(const v8bf*)(arow1 + k + 16);
            const __bf16* bp0 = wlds + (size_t)(mi)      * LDS_STRIDE + k + hi * 8;
            const __bf16* bp1 = wlds + (size_t)(16 + mi) * LDS_STRIDE + k + hi * 8;
            b0.h8[0] = *(const v8bf*)bp0;
            b0.h8[1] = *(const v8bf*)(bp0 + 16);
            b1.h8[0] = *(const v8bf*)bp1;
            b1.h8[1] = *(const v8bf*)(bp1 + 16);
            acc00 = wmma_bf16(a0.v, b0.v, acc00);
            acc01 = wmma_bf16(a0.v, b1.v, acc01);
            acc10 = wmma_bf16(a1.v, b0.v, acc10);
            acc11 = wmma_bf16(a1.v, b1.v, acc11);
        }

        // fused epilogue: h = 0.9*h + 0.1*relu(u + acc + b)
        v8f* accp[2][2] = {{&acc00, &acc01}, {&acc10, &acc11}};
#pragma unroll
        for (int jm = 0; jm < 2; ++jm) {
#pragma unroll
            for (int jn = 0; jn < 2; ++jn) {
                const int n = n_base + jn * 16 + mi;
                const float bias = jn ? bias1 : bias0;
#pragma unroll
                for (int r = 0; r < 8; ++r) {
                    const int m = m0 + jm * 16 + r + hi * 8;
                    const size_t idx = (size_t)m * N_HID + n;
                    const float u = out_t[idx];  // u_t from inp_proj
                    const float v = fmaxf(u + (*accp[jm][jn])[r] + bias, 0.0f);
                    const float hnew = hreg[jm][jn][r] * (1.0f - ALPHA_F) + ALPHA_F * v;
                    hreg[jm][jn][r] = hnew;
                    out_t[idx] = hnew;             // overwrite u_t with h_t
                    hout[idx]  = (__bf16)hnew;     // next step's A operand
                }
            }
        }
        grid_barrier(&bar[0], &bar[1]);  // all h_t visible before step t+1
    }
}

// ---- helpers ----------------------------------------------------------------
__global__ void cvt_f32_to_bf16(const float* __restrict__ s, __bf16* __restrict__ d, int n) {
    int i = blockIdx.x * blockDim.x + threadIdx.x;
    if (i < n) d[i] = (__bf16)s[i];
}

__global__ void zero_barrier(unsigned* __restrict__ bar) {
    if (threadIdx.x < 2) bar[threadIdx.x] = 0u;
}

// ---- launcher ----------------------------------------------------------------
extern "C" void kernel_launch(void* const* d_in, const int* in_sizes, int n_in,
                              void* d_out, int out_size, void* d_ws, size_t ws_size,
                              hipStream_t stream) {
    const float* x    = (const float*)d_in[0];  // [T,B,I]
    const float* W_in = (const float*)d_in[1];  // [H,I]
    const float* b_in = (const float*)d_in[2];  // [H]
    const float* W_hh = (const float*)d_in[3];  // [H,H]
    const float* b_hh = (const float*)d_in[4];  // [H]
    float* out = (float*)d_out;                 // [T,B,H] ++ [B,H]

    // workspace: Wi_bf (512K) | h_bf0 (512K) | h_bf1 (512K) | barrier (256B)
    char* ws = (char*)d_ws;
    __bf16*   Wi_bf = (__bf16*)(ws);
    __bf16*   h_bf0 = (__bf16*)(ws + (512u << 10));
    __bf16*   h_bf1 = (__bf16*)(ws + (1024u << 10));
    unsigned* bar   = (unsigned*)(ws + (1536u << 10));

    zero_barrier<<<1, 64, 0, stream>>>(bar);
    cvt_f32_to_bf16<<<(N_HID * N_IN + 255) / 256, 256, 0, stream>>>(W_in, Wi_bf,
                                                                    N_HID * N_IN);

    // phase 1: inp_proj -> d_out (HBM-bound: ~670 MB => ~29 us at 23.3 TB/s)
    dim3 g1((T_SEQ * N_BATCH) / 128, N_HID / 64);
    ctrnn_inproj<<<g1, 256, 0, stream>>>(x, Wi_bf, b_in, out);

    // phase 2: single persistent kernel, all 512 steps, weights LDS-stationary
    ctrnn_scan<<<STEP_WGS, 256, 0, stream>>>(W_hh, b_hh, out, h_bf0, h_bf1, bar);

    // final hidden state == output[T-1]
    hipMemcpyAsync(out + (size_t)T_SEQ * N_BATCH * N_HID,
                   out + (size_t)(T_SEQ - 1) * N_BATCH * N_HID,
                   (size_t)N_BATCH * N_HID * sizeof(float),
                   hipMemcpyDeviceToDevice, stream);
}
